// Block_57913339019874
// MI455X (gfx1250) — matrix operand order repaired and checked by reading, not compiled, not run on hardware
//
#include <hip/hip_runtime.h>
#include <hip/hip_bf16.h>
#include <stdint.h>

// ---------------- CDNA5 (gfx1250) types & helpers ----------------
typedef __attribute__((ext_vector_type(16))) __bf16 v16bf;
typedef __attribute__((ext_vector_type(8)))  __bf16 v8bf;
typedef __attribute__((ext_vector_type(8)))  float  v8f;

#define DEV static __device__ __forceinline__

DEV v8f wmma_bf16(v16bf a, v16bf b, v8f c) {
  // D = A(16x32) x B(32x16) + C, f32 accum. wave32 only.
  return __builtin_amdgcn_wmma_f32_16x16x32_bf16(false, a, false, b, (short)0, c,
                                                 false, false);
}
DEV v16bf cat8(v8bf lo, v8bf hi) {
  return __builtin_shufflevector(lo, hi, 0,1,2,3,4,5,6,7,8,9,10,11,12,13,14,15);
}

// Problem constants
#define TT 10
#define BB 64
#define CC 512
#define NN 64
#define HH 8
#define TB 640            // T*B
#define HID 2048

// ---------------- weight pre-pack into WMMA A-fragment order ----------------
// A-fragment (16x32 bf16): lane l holds row M=l&15; kb=(l>>4)*8;
// half i -> K = (i<8 ? kb+i : 16+kb+(i-8)).  Packed: [tile][lane][16 halves].
__global__ void pack_w_kernel(const float* __restrict__ W, __bf16* __restrict__ Wp,
                              int Cin, int Cout) {
  int tid = blockIdx.x * blockDim.x + threadIdx.x;
  int total = (Cout >> 4) * (Cin >> 5) * 32;
  if (tid >= total) return;
  int lane = tid & 31;
  int ti   = tid >> 5;
  int ktiles = Cin >> 5;
  int ot = ti / ktiles, kt = ti % ktiles;
  int M  = (ot << 4) + (lane & 15);
  int kb = (lane >> 4) * 8;
  size_t obase = (size_t)ti * 512 + (size_t)lane * 16;
#pragma unroll
  for (int i = 0; i < 16; ++i) {
    int K = (kt << 5) + (i < 8 ? kb + i : 16 + kb + (i - 8));
    Wp[obase + i] = (__bf16)W[(size_t)M * Cin + K];
  }
}

// ---------------- f32 -> bf16 convert ----------------
__global__ void cvt_bf16_kernel(const float* __restrict__ x, __bf16* __restrict__ y,
                                size_t n) {
  size_t i = (size_t)blockIdx.x * blockDim.x + threadIdx.x;
  if (i < n) y[i] = (__bf16)x[i];
}

// ---------------- batched GEMM: Y[tb,o,n] = sum_c W[o,c] X[tb,c,n] ----------
// + fused per-channel sum/sumsq (BN statistics) via atomics.
// grid = (TB, Cout/128), block = 256 (8 waves); wave w -> 16 output rows.
// Double-buffered LDS staging (1 barrier / k-step) + global_prefetch lookahead.
// Staging: 1 coalesced global_load_b128 per thread -> 8x ds_store_b16 scatter.
__global__ void gemm_stats_kernel(const __bf16* __restrict__ X,
                                  const __bf16* __restrict__ Wp,
                                  float* __restrict__ Y,
                                  float* __restrict__ sum,
                                  float* __restrict__ sumsq,
                                  int Cin, int Cout) {
  __shared__ __bf16 lxs[2][64 * 40];       // X tile transposed [n][k], pad 40
  const int tb   = blockIdx.x;
  const int ob   = blockIdx.y * 128;
  const int tid  = threadIdx.x;
  const int wave = tid >> 5, lane = tid & 31;
  const int l15  = lane & 15, lh = lane >> 4;
  const int ktiles = Cin >> 5;
  const int otile  = (ob >> 4) + wave;

  // staging geometry: thread t loads row k=t>>3, cols n0..n0+7 (one b128)
  const int sk = tid >> 3;
  const int sn = (tid & 7) << 3;
  const __bf16* xsrc = X + (size_t)tb * Cin * 64 + (size_t)sk * 64 + sn;
  const __bf16* wsrc = Wp + (size_t)otile * ktiles * 512 + (size_t)lane * 16;

  auto stage = [&](int kt, int buf) {
    v8bf d = *(const v8bf*)(xsrc + (size_t)kt * 2048);
    __bf16* dst = &lxs[buf][sn * 40 + sk];
#pragma unroll
    for (int i = 0; i < 8; ++i) dst[i * 40] = d[i];
  };

  v8f acc[4] = {};
  stage(0, 0);
  __syncthreads();
  for (int kt = 0; kt < ktiles; ++kt) {
    const int cur = kt & 1;
    if (kt + 1 < ktiles) stage(kt + 1, cur ^ 1);
    if (kt + 2 < ktiles) {                  // prefetch 2 tiles ahead
      __builtin_prefetch(xsrc + (size_t)(kt + 2) * 2048, 0, 0);
      __builtin_prefetch(wsrc + (size_t)(kt + 2) * 512, 0, 0);
    }
    // A fragment: one contiguous 32B load per lane from packed weights
    const v16bf a = *(const v16bf*)(wsrc + (size_t)kt * 512);
#pragma unroll
    for (int nt = 0; nt < 4; ++nt) {
      const __bf16* p = &lxs[cur][((nt << 4) + l15) * 40 + (lh << 3)];
      v16bf b = cat8(*(const v8bf*)p, *(const v8bf*)(p + 16));
      acc[nt] = wmma_bf16(a, b, acc[nt]);
    }
    __syncthreads();                        // single barrier: next iter may
  }                                         // overwrite `cur` only after this
  // store + BN stats
  const int ob_w = ob + (wave << 4);
#pragma unroll
  for (int r = 0; r < 8; ++r) {
    int o = ob_w + (lh << 3) + r;
    float s1 = 0.f, s2 = 0.f;
#pragma unroll
    for (int nt = 0; nt < 4; ++nt) {
      float v = acc[nt][r];
      Y[((size_t)tb * Cout + o) * 64 + (nt << 4) + l15] = v;
      s1 += v; s2 += v * v;
    }
    for (int m = 1; m < 16; m <<= 1) {       // reduce over the 16-lane half
      s1 += __shfl_xor(s1, m, 16);
      s2 += __shfl_xor(s2, m, 16);
    }
    if (l15 == 0) {
      atomicAdd(&sum[o], s1);
      atomicAdd(&sumsq[o], s2);
    }
  }
}

// ---------------- BN (optional) + LIF temporal scan ----------------
// mode 0: out[(t*B+b)*C*64 + c*64 + n]   (tb, c, n)
// mode 1: out in head-split layout (t,b,h,nn,c') with h=c&7, nn=c>>3, c'=n
__global__ void bn_lif_kernel(const float* __restrict__ Y,
                              const float* __restrict__ sum,
                              const float* __restrict__ sumsq,
                              const float* __restrict__ gamma,
                              const float* __restrict__ beta,
                              __bf16* __restrict__ out,
                              int C, float thr, int mode, int do_bn) {
  int tid = blockIdx.x * blockDim.x + threadIdx.x;
  int total = BB * C * 64;
  if (tid >= total) return;
  int n = tid & 63;
  int c = (tid >> 6) % C;
  int b = tid / (C * 64);
  float scale = 1.f, shift = 0.f;
  if (do_bn) {
    const float inv = 1.0f / (640.0f * 64.0f);     // TB * N
    float mu  = sum[c] * inv;
    float var = sumsq[c] * inv - mu * mu;
    float rstd = rsqrtf(var + 1e-5f);
    scale = gamma[c] * rstd;
    shift = beta[c] - mu * scale;
  }
  size_t oi;
  if (mode == 0) oi = ((size_t)b * C + c) * 64 + n;
  else {
    int h = c & 7, nn = c >> 3;
    oi = (((size_t)b * 8 + h) * 64 + nn) * 64 + n;
  }
  const size_t ostrideT = (mode == 0) ? (size_t)BB * C * 64 : (size_t)BB * 8 * 64 * 64;
  float mem = 0.f;
#pragma unroll
  for (int t = 0; t < TT; ++t) {
    float z = Y[(((size_t)t * BB + b) * C + c) * 64 + n] * scale + shift;
    mem += (z - mem) * 0.5f;
    float s = (mem > thr) ? 1.f : 0.f;
    mem *= (1.f - s);
    out[(size_t)t * ostrideT + oi] = (__bf16)s;
  }
}

// ---------------- TIM step: kernel-5 conv1d over CoH, LIF(0.3), blend -------
// q layout: (t,b,h,n,c') with c' contiguous; one block per (b,h).
__global__ void tim_step_kernel(__bf16* __restrict__ q,
                                const float* __restrict__ w,
                                const float* __restrict__ bias,
                                float* __restrict__ mem, int t) {
  __shared__ float xs[64][68];
  const int bh = blockIdx.x;                      // b*8 + h
  const int tid = threadIdx.x;
  const size_t prev = ((size_t)(t - 1) * 512 + bh) * 4096;
  const size_t cur  = ((size_t)t * 512 + bh) * 4096;
#pragma unroll
  for (int i = 0; i < 16; ++i) {
    int idx = tid + i * 256;
    xs[idx >> 6][(idx & 63) + 2] = (float)q[prev + idx];
  }
  if (tid < 64) { xs[tid][0] = xs[tid][1] = xs[tid][66] = xs[tid][67] = 0.f; }
  __syncthreads();
  const int n = tid >> 2, jq = tid & 3;
  float acc[16];
#pragma unroll
  for (int jj = 0; jj < 16; ++jj) acc[jj] = bias[n];
  for (int m = 0; m < 64; ++m) {
    const float* wr = &w[(n * 64 + m) * 5];
    float w0 = wr[0], w1 = wr[1], w2 = wr[2], w3 = wr[3], w4 = wr[4];
#pragma unroll
    for (int jj = 0; jj < 16; ++jj) {
      const float* xr = &xs[m][jj * 4 + jq];
      acc[jj] += w0 * xr[0] + w1 * xr[1] + w2 * xr[2] + w3 * xr[3] + w4 * xr[4];
    }
  }
#pragma unroll
  for (int jj = 0; jj < 16; ++jj) {
    int j = jj * 4 + jq;
    size_t mi = (size_t)bh * 4096 + n * 64 + j;
    float mv = mem[mi];
    mv += (acc[jj] - mv) * 0.5f;
    float s = (mv > 0.3f) ? 1.f : 0.f;
    mv *= (1.f - s);
    mem[mi] = mv;
    float qt = (float)q[cur + n * 64 + j];
    q[cur + n * 64 + j] = (__bf16)(0.5f * s + 0.5f * qt);
  }
}

// ---------------- attention: per (t,b,h): attn = q k^T ; o = 0.25 attn v ----
// block = 128 (4 waves); wave -> 16-row strip. Writes merged-head o (f32).
__global__ void attn_kernel(const __bf16* __restrict__ qs,
                            const __bf16* __restrict__ ks,
                            const __bf16* __restrict__ vs,
                            float* __restrict__ o) {
  __shared__ __bf16 vT[64 * 72];       // v transposed [c'][m]
  __shared__ __bf16 at[64 * 72];       // attn bf16 [n][m]
  const int tbh = blockIdx.x;          // (t*64+b)*8+h
  const size_t base = (size_t)tbh * 4096;
  const int tid = threadIdx.x;
#pragma unroll
  for (int i = 0; i < 32; ++i) {
    int idx = tid + i * 128;
    vT[(idx & 63) * 72 + (idx >> 6)] = vs[base + idx];
  }
  __syncthreads();
  const int wave = tid >> 5, lane = tid & 31, l15 = lane & 15, lh = lane >> 4;
  const int nb = wave << 4;
  v8f acc[4] = {};
#pragma unroll
  for (int kk = 0; kk < 2; ++kk) {
    const int c0 = kk << 5;
    const __bf16* qp = qs + base + (size_t)(nb + l15) * 64 + c0 + (lh << 4);
    v16bf a = cat8(*(const v8bf*)qp, *(const v8bf*)(qp + 8));
#pragma unroll
    for (int mt = 0; mt < 4; ++mt) {
      const __bf16* kp = ks + base + (size_t)((mt << 4) + l15) * 64 + c0 + (lh << 4);
      v16bf b = *(const v16bf*)kp;                 // 32B aligned
      acc[mt] = wmma_bf16(a, b, acc[mt]);
    }
  }
#pragma unroll
  for (int mt = 0; mt < 4; ++mt)
#pragma unroll
    for (int r = 0; r < 8; ++r)
      at[(nb + (lh << 3) + r) * 72 + (mt << 4) + l15] = (__bf16)acc[mt][r];
  __syncthreads();
  v8f oc[4] = {};
#pragma unroll
  for (int kk = 0; kk < 2; ++kk) {
    const int m0 = kk << 5;
    const __bf16* ap = &at[(nb + l15) * 72 + m0 + (lh << 4)];
    v16bf a = cat8(*(const v8bf*)ap, *(const v8bf*)(ap + 8));
#pragma unroll
    for (int ct = 0; ct < 4; ++ct) {
      const __bf16* bp = &vT[((ct << 4) + l15) * 72 + m0 + (lh << 4)];
      v16bf b = cat8(*(const v8bf*)bp, *(const v8bf*)(bp + 8));
      oc[ct] = wmma_bf16(a, b, oc[ct]);
    }
  }
  const int tb = tbh >> 3, h = tbh & 7;
#pragma unroll
  for (int ct = 0; ct < 4; ++ct)
#pragma unroll
    for (int r = 0; r < 8; ++r) {
      int cpr = (ct << 4) + l15;
      int n   = nb + (lh << 3) + r;
      o[((size_t)tb * 512 + h * 64 + cpr) * 64 + n] = oc[ct][r] * 0.25f;
    }
}

// ---------------- elementwise residual / convert ----------------
__global__ void add_spike_cvt_kernel(const float* __restrict__ x,
                                     const __bf16* __restrict__ sp,
                                     float* __restrict__ x1,
                                     __bf16* __restrict__ x1b, size_t n) {
  size_t i = (size_t)blockIdx.x * blockDim.x + threadIdx.x;
  if (i >= n) return;
  float v = x[i] + (float)sp[i];
  x1[i] = v;
  x1b[i] = (__bf16)v;
}
__global__ void final_add_kernel(const float* __restrict__ x1,
                                 const __bf16* __restrict__ sp,
                                 float* __restrict__ out, size_t n) {
  size_t i = (size_t)blockIdx.x * blockDim.x + threadIdx.x;
  if (i >= n) return;
  out[i] = x1[i] + (float)sp[i];
}

// ---------------- host ----------------
extern "C" void kernel_launch(void* const* d_in, const int* in_sizes, int n_in,
                              void* d_out, int out_size, void* d_ws, size_t ws_size,
                              hipStream_t stream) {
  const float* x       = (const float*)d_in[0];
  const float* q_w     = (const float*)d_in[1];
  const float* q_gamma = (const float*)d_in[2];
  const float* q_beta  = (const float*)d_in[3];
  const float* k_w     = (const float*)d_in[4];
  const float* k_gamma = (const float*)d_in[5];
  const float* k_beta  = (const float*)d_in[6];
  const float* v_w     = (const float*)d_in[7];
  const float* v_gamma = (const float*)d_in[8];
  const float* v_beta  = (const float*)d_in[9];
  const float* p_w     = (const float*)d_in[10];
  const float* p_gamma = (const float*)d_in[11];
  const float* p_beta  = (const float*)d_in[12];
  const float* tim_w   = (const float*)d_in[13];
  const float* tim_b   = (const float*)d_in[14];
  const float* f1_w    = (const float*)d_in[15];
  // f1_b (16) / f2_b (20): absorbed by BatchNorm mean-subtraction -> unused
  const float* f1_gamma= (const float*)d_in[17];
  const float* f1_beta = (const float*)d_in[18];
  const float* f2_w    = (const float*)d_in[19];
  const float* f2_gamma= (const float*)d_in[21];
  const float* f2_beta = (const float*)d_in[22];
  float* out = (float*)d_out;
  uint8_t* ws = (uint8_t*)d_ws;

  const size_t NX   = (size_t)TT * BB * CC * NN;        // 20,971,520
  const size_t SPL  = NX * 2;                           // bf16 spike buf bytes
  // workspace layout (bytes)
  const size_t o_wq = 0, o_wk = 524288, o_wv = 1048576, o_wp = 1572864;
  const size_t o_wf1 = 2097152, o_wf2 = 4194304;        // 2MB each
  const size_t o_sum = 6291456, o_ssq = o_sum + 8192;   // 2048 f32 each
  const size_t o_xb  = 8388608;
  const size_t o_qs  = o_xb + SPL;
  const size_t o_ks  = o_qs + SPL;
  const size_t o_vs  = o_ks + SPL;
  const size_t o_asp = o_vs + SPL;                      // attn spikes / m spikes
  const size_t o_psp = o_asp + SPL;
  const size_t o_x1  = o_psp + SPL;                     // f32
  const size_t o_x1b = o_x1 + NX * 4;
  const size_t o_hsp = o_x1b + SPL;                     // (tb,2048,n) bf16
  const size_t o_tm  = o_hsp + (size_t)TB * HID * NN * 2;
  const size_t o_Y   = o_tm + (size_t)BB * HH * NN * NN * 4;
  const size_t YB    = (size_t)TB * HID * NN * 4;       // 320MB (max branch)
  if (ws_size < o_Y + YB) return;

  __bf16* xb  = (__bf16*)(ws + o_xb);
  __bf16* qs  = (__bf16*)(ws + o_qs);
  __bf16* ks  = (__bf16*)(ws + o_ks);
  __bf16* vs  = (__bf16*)(ws + o_vs);
  __bf16* asp = (__bf16*)(ws + o_asp);
  __bf16* psp = (__bf16*)(ws + o_psp);
  __bf16* x1b = (__bf16*)(ws + o_x1b);
  __bf16* hsp = (__bf16*)(ws + o_hsp);
  float*  x1  = (float*)(ws + o_x1);
  float*  tmem= (float*)(ws + o_tm);
  float*  Y   = (float*)(ws + o_Y);
  float*  sum = (float*)(ws + o_sum);
  float*  ssq = (float*)(ws + o_ssq);

  // 1) x -> bf16
  cvt_bf16_kernel<<<(NX + 255) / 256, 256, 0, stream>>>(x, xb, NX);
  // 2) pack weights
  auto packs = [&](const float* W, size_t off, int Cin, int Cout) {
    int thr = (Cout * Cin) / 16;
    pack_w_kernel<<<(thr + 255) / 256, 256, 0, stream>>>(W, (__bf16*)(ws + off), Cin, Cout);
  };
  packs(q_w,  o_wq,  CC, CC);  packs(k_w, o_wk, CC, CC);
  packs(v_w,  o_wv,  CC, CC);  packs(p_w, o_wp, CC, CC);
  packs(f1_w, o_wf1, CC, HID); packs(f2_w, o_wf2, HID, CC);

  auto branch = [&](const __bf16* Xin, size_t woff, const float* gamma,
                    const float* beta, __bf16* sp_out, int Cin, int Cout,
                    int mode) {
    hipMemsetAsync(sum, 0, 16384, stream);
    gemm_stats_kernel<<<dim3(TB, Cout / 128), 256, 0, stream>>>(
        Xin, (const __bf16*)(ws + woff), Y, sum, ssq, Cin, Cout);
    int tot = BB * Cout * NN;
    bn_lif_kernel<<<(tot + 255) / 256, 256, 0, stream>>>(
        Y, sum, ssq, gamma, beta, sp_out, Cout, 1.0f, mode, 1);
  };

  // 3) q/k/v branches -> head-split spike layout
  branch(xb, o_wq, q_gamma, q_beta, qs, CC, CC, 1);
  branch(xb, o_wk, k_gamma, k_beta, ks, CC, CC, 1);
  branch(xb, o_wv, v_gamma, v_beta, vs, CC, CC, 1);
  // 4) TIM temporal interaction on q
  hipMemsetAsync(tmem, 0, (size_t)BB * HH * NN * NN * 4, stream);
  for (int t = 1; t < TT; ++t)
    tim_step_kernel<<<BB * HH, 256, 0, stream>>>(qs, tim_w, tim_b, tmem, t);
  // 5) attention -> o (f32, merged heads) into Y region
  attn_kernel<<<TT * BB * HH, 128, 0, stream>>>(qs, ks, vs, Y);
  // 6) attn_lif (thr 0.5, no BN)
  {
    int tot = BB * CC * NN;
    bn_lif_kernel<<<(tot + 255) / 256, 256, 0, stream>>>(
        Y, sum, ssq, q_gamma, q_beta, asp, CC, 0.5f, 0, 0);
  }
  // 7) projection branch, residual
  branch(asp, o_wp, p_gamma, p_beta, psp, CC, CC, 0);
  add_spike_cvt_kernel<<<(NX + 255) / 256, 256, 0, stream>>>(x, psp, x1, x1b, NX);
  // 8) MLP
  branch(x1b, o_wf1, f1_gamma, f1_beta, hsp, CC, HID, 0);
  branch(hsp, o_wf2, f2_gamma, f2_beta, asp, HID, CC, 0);  // reuse asp as m-spikes
  // 9) final residual -> d_out
  final_add_kernel<<<(NX + 255) / 256, 256, 0, stream>>>(x1, asp, out, NX);
}
